// Attention_25271587569919
// MI455X (gfx1250) — compile-verified, matrix operand order checked by reading, not run on hardware
//
#include <hip/hip_runtime.h>

// ---------------- problem constants ----------------
#define NB_HEAD 16
#define DH      64
#define DM      1024            // NB_HEAD * DH
#define BATCH   4
#define SEQ     2048
#define MTOT    (BATCH * SEQ)   // 8192 token rows

typedef __attribute__((ext_vector_type(16))) __bf16 bf16x16;
typedef __attribute__((ext_vector_type(8)))  __bf16 bf16x8;
typedef __attribute__((ext_vector_type(8)))  float  f32x8;

struct U32x8 { uint4 lo, hi; };
static __device__ __forceinline__ bf16x16 mkfrag(uint4 lo, uint4 hi) {
  U32x8 t{lo, hi};
  return __builtin_bit_cast(bf16x16, t);
}

// ---------------- f32 -> bf16 elementwise (8 elems/thread) ----------------
__global__ void __launch_bounds__(256)
cvt_f32_bf16(const float* __restrict__ in, __bf16* __restrict__ out) {
  const size_t i = ((size_t)blockIdx.x * blockDim.x + threadIdx.x) * 8;
  f32x8 a = *(const f32x8*)(in + i);               // 2x global_load_b128
  bf16x8 b = __builtin_convertvector(a, bf16x8);   // native packed cvt
  *(bf16x8*)(out + i) = b;                         // global_store_b128
}

// ---------------- W[K][N] f32 -> WT[N][K] bf16 (transpose+convert) --------
__global__ void __launch_bounds__(256)
wtrans_bf16(const float* __restrict__ W, __bf16* __restrict__ WT) {
  __shared__ float tile[32][33];
  const int n0 = blockIdx.x * 32, k0 = blockIdx.y * 32;
  const int tx = threadIdx.x, ty = threadIdx.y;     // block (32,8)
  #pragma unroll
  for (int r = 0; r < 32; r += 8)
    tile[r + ty][tx] = W[(size_t)(k0 + r + ty) * DM + n0 + tx];
  __syncthreads();
  #pragma unroll
  for (int r = 0; r < 32; r += 8)
    WT[(size_t)(n0 + r + ty) * DM + k0 + tx] = (__bf16)tile[tx][r + ty];
}

// ---------------- tiled WMMA GEMM: C[8192x1024] = A[8192xK] * BT[N][K]^T --
// Block = 8 waves, block tile 256(M) x 64(N); each wave owns 32x64 (8 WMMAs
// per 32-wide K chunk against 2 A-frags + 4 B-frags staged through LDS).
// MODE 0: bf16 out -> [B,H,S,dh]      (Q, K projections)
// MODE 1: bf16 out -> [B,H,dh,S]      (V projection, pre-transposed for PV)
// MODE 2: f32  out -> row-major [M,N] (final output projection)
template <int MODE>
__global__ void __launch_bounds__(256)
gemm_bf16(const __bf16* __restrict__ A,    // [M,1024] row-major bf16
          const __bf16* __restrict__ BT,   // [1024,1024] = B^T row-major bf16
          void* __restrict__ outp) {
  constexpr int K = DM, N = DM;
  __shared__ alignas(16) __bf16 As[256 * 32];   // 16 KB
  __shared__ alignas(16) __bf16 Bs[64 * 32];    //  4 KB
  const int tid = threadIdx.x;
  const int w = tid >> 5, lane = tid & 31;
  const int lane16 = lane & 15, hi = (lane >> 4) & 1;
  const int m0 = blockIdx.y * 256;
  const int n0 = blockIdx.x * 64;

  f32x8 acc[2][4] = {};

  for (int kk = 0; kk < K; kk += 32) {
    // stage A tile 256x32: each thread copies one full 64-byte row
    {
      const uint4* src = (const uint4*)(A + (size_t)(m0 + tid) * K + kk);
      uint4 v0 = src[0], v1 = src[1], v2 = src[2], v3 = src[3];
      if (kk + 32 < K) __builtin_prefetch(A + (size_t)(m0 + tid) * K + kk + 32, 0, 0);
      uint4* dst = (uint4*)(As + tid * 32);
      dst[0] = v0; dst[1] = v1; dst[2] = v2; dst[3] = v3;
    }
    // stage B tile 64x32: each thread copies 16 contiguous bytes
    {
      const int row = tid >> 2, q = (tid & 3) * 8;
      *(uint4*)(Bs + row * 32 + q) =
          *(const uint4*)(BT + (size_t)(n0 + row) * K + kk + q);
      if (kk + 32 < K) __builtin_prefetch(BT + (size_t)(n0 + row) * K + kk + 32 + q, 0, 0);
    }
    __syncthreads();

    bf16x16 af[2];
    #pragma unroll
    for (int i = 0; i < 2; ++i) {
      const int abase = (w * 32 + i * 16 + lane16) * 32 + hi * 8;
      af[i] = mkfrag(*(const uint4*)(As + abase),
                     *(const uint4*)(As + abase + 16));
    }
    #pragma unroll
    for (int j = 0; j < 4; ++j) {
      const int bbase = (j * 16 + lane16) * 32 + hi * 8;
      bf16x16 bfj = mkfrag(*(const uint4*)(Bs + bbase),
                           *(const uint4*)(Bs + bbase + 16));
      #pragma unroll
      for (int i = 0; i < 2; ++i)
        acc[i][j] = __builtin_amdgcn_wmma_f32_16x16x32_bf16(
            false, af[i], false, bfj, (short)0, acc[i][j], false, false);
    }
    __syncthreads();
  }

  // writeback
  #pragma unroll
  for (int i = 0; i < 2; ++i) {
    const int mbase = m0 + w * 32 + i * 16 + hi * 8;
    #pragma unroll
    for (int j = 0; j < 4; ++j) {
      const int n = n0 + j * 16 + lane16;
      #pragma unroll
      for (int r = 0; r < 8; ++r) {
        const int m = mbase + r;
        const float v = acc[i][j][r];
        if (MODE == 0) {
          const int b = m >> 11, s = m & (SEQ - 1);
          const int h = n >> 6, d = n & (DH - 1);
          ((__bf16*)outp)[(((size_t)(b * NB_HEAD + h) * SEQ + s) << 6) + d] = (__bf16)v;
        } else if (MODE == 1) {
          const int b = m >> 11, s = m & (SEQ - 1);
          const int h = n >> 6, d = n & (DH - 1);
          ((__bf16*)outp)[(((size_t)(b * NB_HEAD + h) << 6) + d) * SEQ + s] = (__bf16)v;
        } else {
          ((float*)outp)[(size_t)m * N + n] = v;
        }
      }
    }
  }
}

// ---------------- flash attention (one wave = 16 queries of one head) -----
__global__ void __launch_bounds__(256)
attn_kernel(const __bf16* __restrict__ Qp,  // [B,H,S,64] bf16
            const __bf16* __restrict__ Kp,  // [B,H,S,64] bf16
            const __bf16* __restrict__ VT,  // [B,H,64,S] bf16
            const int* __restrict__ K_ini,  // [B,S]
            __bf16* __restrict__ CTX) {     // [B,S,1024] bf16
  __shared__ alignas(16) __bf16 Plds[8][16 * 32];  // per-wave P tile
  const int tid = threadIdx.x, w = tid >> 5, lane = tid & 31;
  const int lane16 = lane & 15, hi = (lane >> 4) & 1;
  const int b = blockIdx.z, h = blockIdx.y;
  const int q0 = (blockIdx.x * 8 + w) * 16;

  const size_t headoff = (size_t)(b * NB_HEAD + h) * SEQ * DH;
  const __bf16* Qh = Qp + headoff;
  const __bf16* Kh = Kp + headoff;
  const __bf16* Vh = VT + headoff;   // [64][SEQ]
  const int* kini = K_ini + b * SEQ;

  // Q A-fragments, two K-chunks covering dh=64
  bf16x16 aq[2];
  {
    const __bf16* qrow = Qh + (size_t)(q0 + lane16) * DH;
    #pragma unroll
    for (int c = 0; c < 2; ++c) {
      const int base = c * 32 + hi * 8;
      aq[c] = mkfrag(*(const uint4*)(qrow + base),
                     *(const uint4*)(qrow + base + 16));
    }
  }

  f32x8 acc[4] = {};
  float rowmax[8], rowsum[8];
  #pragma unroll
  for (int r = 0; r < 8; ++r) { rowmax[r] = -3.0e38f; rowsum[r] = 0.f; }

  const int nsteps = (q0 + 16 + 31) / 32;   // causal: only keys <= q0+15
  for (int step = 0; step < nsteps; ++step) {
    const int kb = step * 32;

    // scores: two 16x16 subtiles, each = 2 WMMAs over dh
    f32x8 s[2];
    #pragma unroll
    for (int t = 0; t < 2; ++t) {
      const __bf16* krow = Kh + (size_t)(kb + t * 16 + lane16) * DH;
      f32x8 z = {};
      #pragma unroll
      for (int c = 0; c < 2; ++c) {
        const int base = c * 32 + hi * 8;
        bf16x16 kf = mkfrag(*(const uint4*)(krow + base),
                            *(const uint4*)(krow + base + 16));
        z = __builtin_amdgcn_wmma_f32_16x16x32_bf16(
            false, aq[c], false, kf, (short)0, z, false, false);
      }
      s[t] = z;
    }

    // scale 1/sqrt(64) + causal & pad mask (reference: scores - (1-mask)*1e10)
    #pragma unroll
    for (int t = 0; t < 2; ++t) {
      const int key = kb + t * 16 + lane16;
      const int pad = kini[key];
      #pragma unroll
      for (int r = 0; r < 8; ++r) {
        const int mrow = q0 + hi * 8 + r;
        float v = s[t][r] * 0.125f;
        if (key > mrow || pad == 0) v -= 1.0e10f;
        s[t][r] = v;
      }
    }

    // online softmax update (row = 16 lanes within each half-wave)
    float p0[8], p1[8];
    #pragma unroll
    for (int r = 0; r < 8; ++r) {
      float mx = fmaxf(s[0][r], s[1][r]);
      #pragma unroll
      for (int x = 1; x < 16; x <<= 1) mx = fmaxf(mx, __shfl_xor(mx, x, 32));
      const float newmax = fmaxf(rowmax[r], mx);
      const float scale = __expf(rowmax[r] - newmax);
      const float e0 = __expf(s[0][r] - newmax);
      const float e1 = __expf(s[1][r] - newmax);
      float psum = e0 + e1;
      #pragma unroll
      for (int x = 1; x < 16; x <<= 1) psum += __shfl_xor(psum, x, 32);
      rowsum[r] = rowsum[r] * scale + psum;
      rowmax[r] = newmax;
      #pragma unroll
      for (int j = 0; j < 4; ++j) acc[j][r] *= scale;
      p0[r] = e0; p1[r] = e1;
    }

    // P: C-layout f32 -> A-layout bf16 via per-wave LDS (LDS is in-order per wave)
    __bf16* P = Plds[w];
    #pragma unroll
    for (int r = 0; r < 8; ++r) {
      const int rl = hi * 8 + r;
      P[rl * 32 + lane16]      = (__bf16)p0[r];
      P[rl * 32 + 16 + lane16] = (__bf16)p1[r];
    }
    asm volatile("" ::: "memory");   // keep ds_store before ds_load
    {
      const int base = lane16 * 32 + hi * 8;
      bf16x16 pa = mkfrag(*(const uint4*)(P + base),
                          *(const uint4*)(P + base + 16));
      #pragma unroll
      for (int j = 0; j < 4; ++j) {
        const __bf16* vrow = Vh + (size_t)(j * 16 + lane16) * SEQ + kb;
        bf16x16 vf = mkfrag(*(const uint4*)(vrow + hi * 8),
                            *(const uint4*)(vrow + hi * 8 + 16));
        acc[j] = __builtin_amdgcn_wmma_f32_16x16x32_bf16(
            false, pa, false, vf, (short)0, acc[j], false, false);
      }
    }
  }

  // normalize + store ctx as bf16 [B,S,1024] (A-operand of output GEMM)
  #pragma unroll
  for (int r = 0; r < 8; ++r) {
    const float inv = 1.0f / rowsum[r];
    const int srow = q0 + hi * 8 + r;
    const size_t rowoff = ((size_t)b * SEQ + srow) * DM + h * DH;
    #pragma unroll
    for (int j = 0; j < 4; ++j)
      CTX[rowoff + j * 16 + lane16] = (__bf16)(acc[j][r] * inv);
  }
}

// ---------------- host-side orchestration --------------------------------
extern "C" void kernel_launch(void* const* d_in, const int* in_sizes, int n_in,
                              void* d_out, int out_size, void* d_ws, size_t ws_size,
                              hipStream_t stream) {
  (void)in_sizes; (void)n_in; (void)out_size; (void)ws_size;
  const float* Q_emb = (const float*)d_in[0];
  const float* K_emb = (const float*)d_in[1];
  const float* V_emb = (const float*)d_in[2];
  const int*   K_ini = (const int*)d_in[4];   // Q_ini (d_in[3]) unused by reference
  const float* WQ = (const float*)d_in[5];
  const float* WK = (const float*)d_in[6];
  const float* WV = (const float*)d_in[7];
  const float* WO = (const float*)d_in[8];

  const size_t EMB = (size_t)MTOT * DM;   // 8,388,608 elements
  const size_t WEL = (size_t)DM * DM;     // 1,048,576 elements
  char* ws = (char*)d_ws;
  __bf16* Qe  = (__bf16*)ws; ws += EMB * 2;
  __bf16* Ke  = (__bf16*)ws; ws += EMB * 2;
  __bf16* Ve  = (__bf16*)ws; ws += EMB * 2;
  __bf16* WQT = (__bf16*)ws; ws += WEL * 2;
  __bf16* WKT = (__bf16*)ws; ws += WEL * 2;
  __bf16* WVT = (__bf16*)ws; ws += WEL * 2;
  __bf16* WOT = (__bf16*)ws; ws += WEL * 2;
  __bf16* Qp  = (__bf16*)ws; ws += EMB * 2;  // [B,H,S,64]
  __bf16* Kp  = (__bf16*)ws; ws += EMB * 2;  // [B,H,S,64]
  __bf16* VTp = (__bf16*)ws; ws += EMB * 2;  // [B,H,64,S]
  __bf16* CTX = (__bf16*)ws; ws += EMB * 2;  // [B,S,1024]

  const int cvtBlocks = (int)(EMB / (8 * 256));
  cvt_f32_bf16<<<cvtBlocks, 256, 0, stream>>>(Q_emb, Qe);
  cvt_f32_bf16<<<cvtBlocks, 256, 0, stream>>>(K_emb, Ke);
  cvt_f32_bf16<<<cvtBlocks, 256, 0, stream>>>(V_emb, Ve);

  dim3 tb(32, 8), tg(DM / 32, DM / 32);
  wtrans_bf16<<<tg, tb, 0, stream>>>(WQ, WQT);
  wtrans_bf16<<<tg, tb, 0, stream>>>(WK, WKT);
  wtrans_bf16<<<tg, tb, 0, stream>>>(WV, WVT);
  wtrans_bf16<<<tg, tb, 0, stream>>>(WO, WOT);

  dim3 gg(DM / 64, MTOT / 256);
  gemm_bf16<0><<<gg, 256, 0, stream>>>(Qe, WQT, (void*)Qp);
  gemm_bf16<0><<<gg, 256, 0, stream>>>(Ke, WKT, (void*)Kp);
  gemm_bf16<1><<<gg, 256, 0, stream>>>(Ve, WVT, (void*)VTp);

  dim3 ag(SEQ / (16 * 8), NB_HEAD, BATCH);
  attn_kernel<<<ag, 256, 0, stream>>>(Qp, Kp, VTp, K_ini, CTX);

  gemm_bf16<2><<<gg, 256, 0, stream>>>(CTX, WOT, d_out);
}
